// Cross_Attention_2972117369301
// MI455X (gfx1250) — compile-verified, matrix operand order checked by reading, not run on hardware
//
#include <hip/hip_runtime.h>
#include <hip/hip_bf16.h>
#include <math.h>

typedef __attribute__((ext_vector_type(16))) __bf16 v16bf;
typedef __attribute__((ext_vector_type(8)))  float  v8f;

#define HEADS 8
#define B_    8
#define N_    4096
#define D_    512
#define E_    1024
#define KSPLIT 8

// K-index mapping for 16-bit A/B fragments of V_WMMA_*_16X16X32_* (wave32):
// lanes 0-15 hold K={0..7,16..23}, lanes 16-31 hold K={8..15,24..31},
// packed 2 per VGPR (CDNA5 ISA, "16-bit A-Matrix 16x32").
__device__ __forceinline__ int k_of(int e, int khalf) {
  int j = e >> 1, s = e & 1;
  int base = (j < 4) ? (j * 2) : (16 + (j - 4) * 2);
  return base + s + khalf * 8;
}

__device__ __forceinline__ unsigned int pack2bf(float a, float b) {
  unsigned short lo = __builtin_bit_cast(unsigned short, (__bf16)a);
  unsigned short hi = __builtin_bit_cast(unsigned short, (__bf16)b);
  return (unsigned int)lo | ((unsigned int)hi << 16);
}

// ---------------- Pass A: key-softmax denominators -------------------------
// rowsum[b, c] = sum_n exp(x2[b, n, c])   (inputs ~N(0,1): no max-shift needed)
__global__ __launch_bounds__(256) void krn_rowsum(const float* __restrict__ x2,
                                                  float* __restrict__ rowsum) {
  int b = blockIdx.x >> 4;
  int chunk = blockIdx.x & 15;
  int tid = threadIdx.x;
  const float* px = x2 + (size_t)b * N_ * D_ + (size_t)chunk * 256 * D_;
  float a0 = 0.f, a1 = 0.f;
  int c0 = tid, c1 = tid + 256;
  for (int n = 0; n < 256; ++n) {
    a0 += __expf(px[(size_t)n * D_ + c0]);
    a1 += __expf(px[(size_t)n * D_ + c1]);
  }
  atomicAdd(&rowsum[b * D_ + c0], a0);
  atomicAdd(&rowsum[b * D_ + c1], a1);
}

// ---------------- Pass B: context[b,h,k,v] = sum_n key * v -----------------
// One WG per (b, h, k-split). Both LDS tiles K(=token)-major, padded pitch 72.
__global__ __launch_bounds__(256) void krn_context(const float* __restrict__ x1,
                                                   const float* __restrict__ x2,
                                                   const float* __restrict__ rowsum,
                                                   float* __restrict__ context) {
  int ks = blockIdx.x % KSPLIT;
  int bh = blockIdx.x / KSPLIT;
  int h = bh & 7, b = bh >> 3;
  __shared__ __bf16 keyT[64][72];  // A: [k-channel][token]
  __shared__ __bf16 vT[64][72];    // B: [v-channel][token] (K-major)
  int tid = threadIdx.x;
  int wave = tid >> 5, lane = tid & 31;
  int cch = tid & 63;
  int trow = tid >> 6;  // 0..3
  float inv = 1.0f / rowsum[b * D_ + h * 64 + cch];
  const float* x2p = x2 + (size_t)b * N_ * D_ + h * 64;
  const float* x1p = x1 + (size_t)b * N_ * D_ + h * 64;
  v8f acc0 = {}, acc1 = {};
  int tile0 = wave * 2, tile1 = wave * 2 + 1;
  int tm0 = (tile0 >> 2) * 16, tn0 = (tile0 & 3) * 16;
  int tm1 = (tile1 >> 2) * 16, tn1 = (tile1 & 3) * 16;
  int mi = lane & 15, khalf = lane >> 4;
  for (int chunk = 0; chunk < (N_ / KSPLIT / 64); ++chunk) {
    int n0 = ks * (N_ / KSPLIT) + chunk * 64;
    __syncthreads();
#pragma unroll
    for (int p = 0; p < 8; ++p) {                 // 2 consecutive tokens/thread
      int n = p * 8 + trow * 2;
      float k0 = __expf(x2p[(size_t)(n0 + n) * D_ + cch]) * inv;
      float k1 = __expf(x2p[(size_t)(n0 + n + 1) * D_ + cch]) * inv;
      *reinterpret_cast<unsigned int*>(&keyT[cch][n]) = pack2bf(k0, k1);
      float v0 = x1p[(size_t)(n0 + n) * D_ + cch];
      float v1 = x1p[(size_t)(n0 + n + 1) * D_ + cch];
      *reinterpret_cast<unsigned int*>(&vT[cch][n]) = pack2bf(v0, v1);
    }
    __syncthreads();
#pragma unroll
    for (int kc = 0; kc < 64; kc += 32) {
      v16bf a0v, a1v, b0v, b1v;
#pragma unroll
      for (int e = 0; e < 16; ++e) {
        int kk = kc + k_of(e, khalf);
        a0v[e] = keyT[tm0 + mi][kk];
        b0v[e] = vT[tn0 + mi][kk];
        a1v[e] = keyT[tm1 + mi][kk];
        b1v[e] = vT[tn1 + mi][kk];
      }
      acc0 = __builtin_amdgcn_wmma_f32_16x16x32_bf16(false, a0v, false, b0v,
                                                     (short)0, acc0, false, false);
      acc1 = __builtin_amdgcn_wmma_f32_16x16x32_bf16(false, a1v, false, b1v,
                                                     (short)0, acc1, false, false);
    }
  }
  // C/D layout: VGPR r, lane L -> M = r + 8*(L>=16), N = L%16
  float* ctx = context + ((size_t)b * HEADS + h) * 64 * 64;
  int nn = lane & 15, mh = lane >> 4;
#pragma unroll
  for (int r = 0; r < 8; ++r) {
    int m = r + mh * 8;
    atomicAdd(&ctx[(tm0 + m) * 64 + tn0 + nn], acc0[r]);
    atomicAdd(&ctx[(tm1 + m) * 64 + tn1 + nn], acc1[r]);
  }
}

// ---------------- Pass C: W2[b, e, h*64+k] = sum_v ctx[k,v] * conv_w[e, h*64+v]
// Output layout [b][e][c] (c contiguous) so pass D stages K-major trivially.
__global__ __launch_bounds__(256) void krn_w2(const float* __restrict__ context,
                                              const float* __restrict__ conv_w,
                                              unsigned short* __restrict__ W2u) {
  __bf16* W2 = reinterpret_cast<__bf16*>(W2u);
  int h = blockIdx.x & 7, b = blockIdx.x >> 3;
  __shared__ __bf16 ctxT[64][72];   // A: [k][v]        (K(=v)-contiguous)
  __shared__ __bf16 wT[128][72];    // B: [e][v]        (K-major)
  int tid = threadIdx.x, wave = tid >> 5, lane = tid & 31;
  const float* ctx = context + ((size_t)b * HEADS + h) * 64 * 64;
  for (int i = tid; i < 2048; i += 256) {         // packed pair stores
    int k = i >> 5, v2 = (i & 31) * 2;
    *reinterpret_cast<unsigned int*>(&ctxT[k][v2]) =
        pack2bf(ctx[k * 64 + v2], ctx[k * 64 + v2 + 1]);
  }
  int mi = lane & 15, khalf = lane >> 4;
  __bf16* w2p = W2 + (size_t)b * E_ * D_;
  for (int eb = 0; eb < E_; eb += 128) {
    __syncthreads();
    for (int i = tid; i < 2048; i += 256) {       // float4 -> 4x bf16
      int e = i >> 4, v4 = (i & 15) * 4;
      const float4 f = *reinterpret_cast<const float4*>(
          conv_w + (size_t)(eb + e) * D_ + h * 64 + v4);
      ushort4 u;
      u.x = __builtin_bit_cast(unsigned short, (__bf16)f.x);
      u.y = __builtin_bit_cast(unsigned short, (__bf16)f.y);
      u.z = __builtin_bit_cast(unsigned short, (__bf16)f.z);
      u.w = __builtin_bit_cast(unsigned short, (__bf16)f.w);
      *reinterpret_cast<ushort4*>(&wT[e][v4]) = u;
    }
    __syncthreads();
#pragma unroll
    for (int t = 0; t < 4; ++t) {
      int tile = wave * 4 + t;          // 32 tiles: 4 (k) x 8 (e)
      int tm = (tile >> 3) * 16;
      int tn = (tile & 7) * 16;
      v8f acc = {};
#pragma unroll
      for (int kc = 0; kc < 64; kc += 32) {
        v16bf a, bb;
#pragma unroll
        for (int e = 0; e < 16; ++e) {
          int kk = kc + k_of(e, khalf);
          a[e] = ctxT[tm + mi][kk];
          bb[e] = wT[tn + mi][kk];
        }
        acc = __builtin_amdgcn_wmma_f32_16x16x32_bf16(false, a, false, bb,
                                                      (short)0, acc, false, false);
      }
      // store: lane owns 8 consecutive c at row e -> single uint4 store
      int nn = lane & 15, mh = lane >> 4;
      uint4 w;
      w.x = pack2bf(acc[0], acc[1]);
      w.y = pack2bf(acc[2], acc[3]);
      w.z = pack2bf(acc[4], acc[5]);
      w.w = pack2bf(acc[6], acc[7]);
      __bf16* dst = w2p + (size_t)(eb + tn + nn) * D_ + h * 64 + tm + mh * 8;
      *reinterpret_cast<uint4*>(dst) = w;
    }
  }
}

// ---------------- Pass D: query softmax + main GEMM + bias + LayerNorm -----
// y[b, n0+t, e] = LN( sum_c query[t,c] * W2[b,e,c] + conv_b[e] )
__global__ __launch_bounds__(256) void krn_fused(const float* __restrict__ x2,
                                                 const unsigned short* __restrict__ W2u,
                                                 const float* __restrict__ conv_b,
                                                 const float* __restrict__ ln_w,
                                                 const float* __restrict__ ln_b,
                                                 float* __restrict__ out) {
  const __bf16* W2 = reinterpret_cast<const __bf16*>(W2u);
  int b = blockIdx.x >> 8;
  int n0 = (blockIdx.x & 255) * 16;
  int tid = threadIdx.x, wave = tid >> 5, lane = tid & 31;

  __shared__ __bf16 qTile[16][520];                   // A tile, padded pitch
  __shared__ __align__(16) unsigned char big[81920];  // w2T / x2s / ypre
  __shared__ float smax[16][HEADS], ssum[16][HEADS];
  __shared__ float ps[16][16], pq[16][16];
  __shared__ float mu_s[16], rs_s[16];

  // 1) stage x2 tile (contiguous 32 KB copy)
  float* x2s = reinterpret_cast<float*>(big);
  const float* x2g = x2 + ((size_t)b * N_ + n0) * D_;
  for (int i = tid; i < 16 * D_; i += 256) x2s[i] = x2g[i];
  __syncthreads();
  // 2) per (token, head) softmax stats over the 64 head channels
  if (tid < 16 * HEADS) {
    int t = tid >> 3, h = tid & 7;
    const float* row = x2s + t * D_ + h * 64;
    float m = -1e30f;
    for (int j = 0; j < 64; ++j) m = fmaxf(m, row[j]);
    float s = 0.f;
    for (int j = 0; j < 64; ++j) s += __expf(row[j] - m);
    smax[t][h] = m;
    ssum[t][h] = 1.0f / s;
  }
  __syncthreads();
  // 3) bf16 query tile (packed pair stores)
  for (int i = tid; i < 8 * D_; i += 256) {
    int t = i >> 8, c = (i & 255) * 2;
    int h = c >> 6;
    float q0 = __expf(x2s[t * D_ + c] - smax[t][h]) * ssum[t][h];
    float q1 = __expf(x2s[t * D_ + c + 1] - smax[t][h]) * ssum[t][h];
    *reinterpret_cast<unsigned int*>(&qTile[t][c]) = pack2bf(q0, q1);
  }
  __syncthreads();

  // 4) GEMM [16 x 512] x [512 x 1024]; wave w covers e in [w*128, w*128+128)
  __bf16 (*w2T)[40] = reinterpret_cast<__bf16 (*)[40]>(big);  // [e][32c], pad 40
  v8f acc[8] = {};
  int mi = lane & 15, khalf = lane >> 4;
  const __bf16* W2b = W2 + (size_t)b * E_ * D_;
  for (int kc0 = 0; kc0 < D_; kc0 += 32) {
    __syncthreads();
    for (int i = tid; i < 4 * E_; i += 256) {   // uint4 row-segment copies
      int e = i >> 2, seg = i & 3;
      *reinterpret_cast<uint4*>(&w2T[e][seg * 8]) =
          *reinterpret_cast<const uint4*>(W2b + (size_t)e * D_ + kc0 + seg * 8);
    }
    if (kc0 + 32 < D_)  // hint next chunk toward cache (global_prefetch_b8)
      __builtin_prefetch(W2b + (size_t)(tid * 4) * D_ + kc0 + 32, 0, 1);
    __syncthreads();
    v16bf a;
#pragma unroll
    for (int e = 0; e < 16; ++e) a[e] = qTile[mi][kc0 + k_of(e, khalf)];
#pragma unroll
    for (int t = 0; t < 8; ++t) {
      int tn = (wave * 8 + t) * 16;
      v16bf bb;
#pragma unroll
      for (int e = 0; e < 16; ++e) bb[e] = w2T[tn + mi][k_of(e, khalf)];
      acc[t] = __builtin_amdgcn_wmma_f32_16x16x32_bf16(false, a, false, bb,
                                                       (short)0, acc[t], false, false);
    }
  }
  __syncthreads();

  // 5) epilogue: bias, then LayerNorm over e (reuse big as f32 [16][1032])
  float (*ypre)[1032] = reinterpret_cast<float (*)[1032]>(big);
  int nn = lane & 15, mh = lane >> 4;
#pragma unroll
  for (int t = 0; t < 8; ++t) {
    int tn = (wave * 8 + t) * 16;
#pragma unroll
    for (int r = 0; r < 8; ++r) {
      int m = r + mh * 8;
      ypre[m][tn + nn] = acc[t][r] + conv_b[tn + nn];
    }
  }
  __syncthreads();
  {
    int t = tid >> 4, seg = tid & 15;
    float s = 0.f, q = 0.f;
    for (int j = 0; j < 64; ++j) {            // stride-16: bank-conflict free
      float v = ypre[t][seg + j * 16];
      s += v; q += v * v;
    }
    ps[t][seg] = s; pq[t][seg] = q;
  }
  __syncthreads();
  if (tid < 16) {
    float s = 0.f, q = 0.f;
    for (int j = 0; j < 16; ++j) { s += ps[tid][j]; q += pq[tid][j]; }
    float mu = s * (1.0f / E_);
    float var = q * (1.0f / E_) - mu * mu;
    mu_s[tid] = mu;
    rs_s[tid] = rsqrtf(var + 1e-5f);
  }
  __syncthreads();
  float* outp = out + ((size_t)b * N_ + n0) * E_;
  for (int i = tid; i < 16 * E_; i += 256) {
    int t = i >> 10, e = i & (E_ - 1);
    outp[(size_t)t * E_ + e] = (ypre[t][e] - mu_s[t]) * rs_s[t] * ln_w[e] + ln_b[e];
  }
}

extern "C" void kernel_launch(void* const* d_in, const int* in_sizes, int n_in,
                              void* d_out, int out_size, void* d_ws, size_t ws_size,
                              hipStream_t stream) {
  (void)in_sizes; (void)n_in; (void)out_size; (void)ws_size;
  const float* x1     = (const float*)d_in[0];
  const float* x2     = (const float*)d_in[1];
  const float* conv_w = (const float*)d_in[2];
  const float* conv_b = (const float*)d_in[3];
  const float* ln_w   = (const float*)d_in[4];
  const float* ln_b   = (const float*)d_in[5];
  float* out = (float*)d_out;

  char* ws = (char*)d_ws;
  float* rowsum      = (float*)ws;                              // 16 KB
  float* context     = (float*)(ws + 16384);                    // 1 MB
  unsigned short* W2 = (unsigned short*)(ws + 16384 + 1048576); // 8 MB bf16

  // zero the atomic-accumulation regions (rowsum + context), graph-capture safe
  hipMemsetAsync(ws, 0, 16384 + 1048576, stream);

  krn_rowsum <<<B_ * 16,             256, 0, stream>>>(x2, rowsum);
  krn_context<<<B_ * HEADS * KSPLIT, 256, 0, stream>>>(x1, x2, rowsum, context);
  krn_w2     <<<B_ * HEADS,          256, 0, stream>>>(context, conv_w, W2);
  krn_fused  <<<B_ * (N_ / 16),      256, 0, stream>>>(x2, W2, conv_b, ln_w, ln_b, out);
}
